// Elman_12137577578506
// MI455X (gfx1250) — matrix operand order repaired and checked by reading, not compiled, not run on hardware
//
#include <hip/hip_runtime.h>
#include <math.h>

// ---------------------------------------------------------------------------
// Elman RNN with SiLU output gating, persistent-kernel formulation for
// MI455X (gfx1250). bf16 WMMA 16x16x32, fp32 accumulate. Weights stay in the
// 192MB L2; per-step dependency handled with device-wide generation barriers.
// Phase schedule (3 barriers/step):
//   A: all 128 WGs:    hidden = tanh([x;h] @ W1^T)          (16 WMMA/wave)
//   B: WGs 0-63:       h_new  = hidden @ W2^T               (16 WMMA/wave)
//      WGs 64-127:     zx     = Wgx @ x_t   (indep. of B!)  ( 8 WMMA/wave)
//   C: WGs 0-63:       z = Wgh @ h_new + zx + bias; out = h_new * silu(z)
//                                                           ( 8 WMMA/wave)
//      WGs 64-127:     stage x[t+1] -> bf16 concat buffer
// ---------------------------------------------------------------------------

typedef __attribute__((ext_vector_type(16))) __bf16 v16bf;
typedef __attribute__((ext_vector_type(8)))  float  v8f;
typedef unsigned short bfu;   // raw bf16 storage

constexpr int Dd = 1024;   // hidden size
constexpr int Ee = 2048;   // expanded size
constexpr int Tt = 1024;   // timesteps
constexpr int Bb = 16;     // batch (== WMMA M)
constexpr int NWG  = 128;  // persistent workgroups
constexpr int NTHR = 128;  // 4 wave32 per WG

// workspace layout (bytes, all >=16B aligned)
constexpr size_t OFF_SYNC  = 0;                              // 2 uints
constexpr size_t OFF_ACAT0 = 256;                            // [16,2048] bf16 = [x_t ; h]
constexpr size_t OFF_ACAT1 = OFF_ACAT0 + (size_t)Bb*Ee*2;    // double buffer
constexpr size_t OFF_HID   = OFF_ACAT1 + (size_t)Bb*Ee*2;    // hidden [16,2048] bf16
constexpr size_t OFF_HNF   = OFF_HID   + (size_t)Bb*Ee*2;    // h_new  [16,1024] f32
constexpr size_t OFF_HNB   = OFF_HNF   + (size_t)Bb*Dd*4;    // h_new  [16,1024] bf16
constexpr size_t OFF_ZX    = OFF_HNB   + (size_t)Bb*Dd*2;    // Wgx@x  [16,1024] f32
constexpr size_t OFF_W1    = OFF_ZX    + (size_t)Bb*Dd*4;    // [2048,2048] bf16
constexpr size_t OFF_W2    = OFF_W1    + (size_t)Ee*2*Dd*2;  // [1024,2048] bf16
constexpr size_t OFF_WGX   = OFF_W2    + (size_t)Dd*Ee*2;    // [1024,1024] bf16
constexpr size_t OFF_WGH   = OFF_WGX   + (size_t)Dd*Dd*2;    // [1024,1024] bf16

__device__ __host__ inline unsigned short f2bf_bits(float f) {
    union { float f; unsigned u; } v; v.f = f;
    unsigned r = (v.u + 0x7FFFu + ((v.u >> 16) & 1u)) >> 16;  // RNE
    return (unsigned short)r;
}

// A-operand per-lane fetch: row = lane%16; VGPR0-3 <- K[8h .. 8h+8),
// VGPR4-7 <- K[16+8h .. 16+8h+8)  (ISA 7.12.2, 16-bit A 16x32)
__device__ inline v16bf ldA(const bfu* p) {
    union { uint4 u[2]; v16bf v; } t;
    t.u[0] = *(const uint4*)(p);
    t.u[1] = *(const uint4*)(p + 16);
    return t.v;
}
// B-operand per-lane fetch: col = lane%16; VGPR0-3 <- K[8h .. 8h+8),
// VGPR4-7 <- K[16+8h .. 16+8h+8)  (same K order as A)
__device__ inline v16bf ldB(const bfu* p) {
    union { uint4 u[2]; v16bf v; } t;
    t.u[0] = *(const uint4*)(p);
    t.u[1] = *(const uint4*)(p + 16);
    return t.v;
}

// C += A[16 x klen] * B^T, K-chunk starting at k0, strides in elements.
__device__ inline v8f gemm_chunk(const bfu* Abase, int as,
                                 const bfu* Bbase, int bs,
                                 int k0, int klen, v8f c, int lane) {
    const int r = lane & 15;
    const int h = (lane >> 4) & 1;
    const bfu* ap = Abase + (size_t)r * as + k0 + 8 * h;
    const bfu* bp = Bbase + (size_t)r * bs + k0 + 8 * h;
    __builtin_prefetch(bp + 256, 0, 0);   // global_prefetch_b8 into L2/WGP$
#pragma unroll 4
    for (int kk = 0; kk < klen; kk += 32) {
        v16bf a = ldA(ap + kk);
        v16bf b = ldB(bp + kk);
        c = __builtin_amdgcn_wmma_f32_16x16x32_bf16(
                false, a, false, b, (short)0, c, false, false);
    }
    return c;
}

// device-wide generation barrier (all NWG workgroups resident)
__device__ inline void gbar(unsigned* sync) {
    __syncthreads();
    if (threadIdx.x == 0) {
        __threadfence();
        unsigned gen = atomicAdd(&sync[1], 0u);
        if (atomicAdd(&sync[0], 1u) + 1u == (unsigned)NWG) {
            atomicExch(&sync[0], 0u);
            __threadfence();
            atomicAdd(&sync[1], 1u);
        } else {
            while (atomicAdd(&sync[1], 0u) == gen) {
                __builtin_amdgcn_s_sleep(1);
            }
        }
        __threadfence();
    }
    __syncthreads();
}

__global__ void init_sync_kernel(unsigned* s) { s[0] = 0u; s[1] = 0u; }

__global__ void cvt_bf16_kernel(const float* __restrict__ src,
                                bfu* __restrict__ dst, int n) {
    for (int i = blockIdx.x * blockDim.x + threadIdx.x; i < n;
         i += gridDim.x * blockDim.x)
        dst[i] = f2bf_bits(src[i]);
}

__global__ __launch_bounds__(NTHR, 1)
void elman_persistent_kernel(const float* __restrict__ x,
                             const float* __restrict__ h0,
                             const float* __restrict__ bias,
                             float* __restrict__ out,
                             char* __restrict__ ws) {
    const int tid  = threadIdx.x;
    const int lane = tid & 31;
    const int wave = tid >> 5;
    const int wg   = blockIdx.x;

    unsigned* sync = (unsigned*)(ws + OFF_SYNC);
    bfu* acat0 = (bfu*)(ws + OFF_ACAT0);
    bfu* acat1 = (bfu*)(ws + OFF_ACAT1);
    bfu* hid   = (bfu*)(ws + OFF_HID);
    float* hnf = (float*)(ws + OFF_HNF);
    bfu* hnb   = (bfu*)(ws + OFF_HNB);
    float* zx  = (float*)(ws + OFF_ZX);
    const bfu* W1b  = (const bfu*)(ws + OFF_W1);
    const bfu* W2b  = (const bfu*)(ws + OFF_W2);
    const bfu* Wgxb = (const bfu*)(ws + OFF_WGX);
    const bfu* Wghb = (const bfu*)(ws + OFF_WGH);

    __shared__ float red[4][8][32];

    // stage [x_0 ; h0] into acat0 (one element of each per thread)
    {
        int idx = wg * NTHR + tid;        // 0 .. 16383 == B*D
        int b = idx >> 10, d = idx & 1023;
        acat0[(size_t)b * Ee + d]      = f2bf_bits(x[idx]);
        acat0[(size_t)b * Ee + Dd + d] = f2bf_bits(h0[idx]);
    }
    gbar(sync);

    for (int t = 0; t < Tt; ++t) {
        const bfu* A0 = (t & 1) ? acat1 : acat0;   // current [x_t ; h]
        bfu*       An = (t & 1) ? acat0 : acat1;   // staged for t+1

        // ---- Phase A: hidden = tanh([x;h] @ W1^T), 128 tiles of E --------
        {
            v8f c = {};
            c = gemm_chunk(A0, Ee, W1b + (size_t)wg * 16 * (2 * Dd), 2 * Dd,
                           wave * 512, 512, c, lane);
#pragma unroll
            for (int r = 0; r < 8; ++r) red[wave][r][lane] = c[r];
            __syncthreads();
            if (wave == 0) {
                int half = lane >> 4, col = wg * 16 + (lane & 15);
#pragma unroll
                for (int r = 0; r < 8; ++r) {
                    float s = red[0][r][lane] + red[1][r][lane] +
                              red[2][r][lane] + red[3][r][lane];
                    int b = r + half * 8;
                    hid[(size_t)b * Ee + col] = f2bf_bits(tanhf(s));
                }
            }
        }
        gbar(sync);

        // ---- Phase B: WGs 0-63:  h_new = hidden @ W2^T (16 WMMA/wave)
        //               WGs 64-127: zx = Wgx @ x_t       ( 8 WMMA/wave) ----
        if (wg < 64) {
            v8f c = {};
            c = gemm_chunk(hid, Ee, W2b + (size_t)wg * 16 * Ee, Ee,
                           wave * 512, 512, c, lane);
#pragma unroll
            for (int r = 0; r < 8; ++r) red[wave][r][lane] = c[r];
            __syncthreads();
            if (wave == 0) {
                int half = lane >> 4, col = wg * 16 + (lane & 15);
#pragma unroll
                for (int r = 0; r < 8; ++r) {
                    float s = red[0][r][lane] + red[1][r][lane] +
                              red[2][r][lane] + red[3][r][lane];
                    int b = r + half * 8;
                    hnf[(size_t)b * Dd + col] = s;
                    hnb[(size_t)b * Dd + col] = f2bf_bits(s);
                }
            }
        } else {
            const int tile = wg - 64;
            v8f c = {};
            // x_t lives in cols 0..1023 of A0 (stride Ee); independent of B!
            c = gemm_chunk(A0, Ee, Wgxb + (size_t)tile * 16 * Dd, Dd,
                           wave * 256, 256, c, lane);
#pragma unroll
            for (int r = 0; r < 8; ++r) red[wave][r][lane] = c[r];
            __syncthreads();
            if (wave == 0) {
                int half = lane >> 4, col = tile * 16 + (lane & 15);
#pragma unroll
                for (int r = 0; r < 8; ++r) {
                    float s = red[0][r][lane] + red[1][r][lane] +
                              red[2][r][lane] + red[3][r][lane];
                    int b = r + half * 8;
                    zx[(size_t)b * Dd + col] = s;
                }
            }
        }
        gbar(sync);

        // ---- Phase C: z = Wgh@h_new + zx + bias; out = h_new * silu(z) ---
        if (wg < 64) {
            v8f c = {};
            c = gemm_chunk(hnb, Dd, Wghb + (size_t)wg * 16 * Dd, Dd,
                           wave * 256, 256, c, lane);
#pragma unroll
            for (int r = 0; r < 8; ++r) red[wave][r][lane] = c[r];
            __syncthreads();
            if (wave == 0) {
                int half = lane >> 4, col = wg * 16 + (lane & 15);
                float bb = bias[col];
#pragma unroll
                for (int r = 0; r < 8; ++r) {
                    int b = r + half * 8;
                    float z = red[0][r][lane] + red[1][r][lane] +
                              red[2][r][lane] + red[3][r][lane] +
                              zx[(size_t)b * Dd + col] + bb;
                    float gate = z / (1.0f + __expf(-z));   // silu(z)
                    float h = hnf[(size_t)b * Dd + col];
                    float o = h * gate;
                    out[(size_t)t * (Bb * Dd) + (size_t)b * Dd + col] = o;
                    An[(size_t)b * Ee + Dd + col] = f2bf_bits(o);  // carried h
                }
            }
        } else {
            // WGs 64..127: stage x[t+1] into the other buffer concurrently
            if (t + 1 < Tt) {
                const float* xt = x + (size_t)(t + 1) * (Bb * Dd);
                int base = (wg - 64) * 256 + tid * 2;
#pragma unroll
                for (int j = 0; j < 2; ++j) {
                    int idx = base + j;
                    int b = idx >> 10, d = idx & 1023;
                    An[(size_t)b * Ee + d] = f2bf_bits(xt[idx]);
                }
            }
        }
        gbar(sync);
    }
}

extern "C" void kernel_launch(void* const* d_in, const int* in_sizes, int n_in,
                              void* d_out, int out_size, void* d_ws, size_t ws_size,
                              hipStream_t stream) {
    const float* x    = (const float*)d_in[0];
    const float* h0   = (const float*)d_in[1];
    const float* W1   = (const float*)d_in[2];
    const float* W2   = (const float*)d_in[3];
    const float* Wgx  = (const float*)d_in[4];
    const float* Wgh  = (const float*)d_in[5];
    const float* bias = (const float*)d_in[6];
    char* ws = (char*)d_ws;

    // reset barrier state every call (ws is not re-poisoned between replays)
    init_sync_kernel<<<1, 1, 0, stream>>>((unsigned*)(ws + OFF_SYNC));

    // fp32 -> bf16 weight staging (weights then live in L2 for the scan)
    cvt_bf16_kernel<<<2048, 256, 0, stream>>>(W1,  (bfu*)(ws + OFF_W1),  Ee * 2 * Dd);
    cvt_bf16_kernel<<<2048, 256, 0, stream>>>(W2,  (bfu*)(ws + OFF_W2),  Dd * Ee);
    cvt_bf16_kernel<<<1024, 256, 0, stream>>>(Wgx, (bfu*)(ws + OFF_WGX), Dd * Dd);
    cvt_bf16_kernel<<<1024, 256, 0, stream>>>(Wgh, (bfu*)(ws + OFF_WGH), Dd * Dd);

    elman_persistent_kernel<<<NWG, NTHR, 0, stream>>>(
        x, h0, bias, (float*)d_out, ws);
}
